// PointRefineLayer_26456998544021
// MI455X (gfx1250) — compile-verified, matrix operand order checked by reading, not run on hardware
//
#include <hip/hip_runtime.h>
#include <cstddef>
#include <cstdint>

// ---------------------------------------------------------------------------
// CDNA5 (gfx1250) WMMA types
// ---------------------------------------------------------------------------
typedef __bf16 bf16_t;
typedef bf16_t bf16x16 __attribute__((ext_vector_type(16)));
typedef unsigned short u16x16 __attribute__((ext_vector_type(16)));
typedef unsigned short u16x8 __attribute__((ext_vector_type(8)));   // 16B, LDS b128
typedef unsigned short u16x4 __attribute__((ext_vector_type(4)));   // 8B, LDS b64
typedef float f32x8 __attribute__((ext_vector_type(8)));

#define ACT_NONE 0
#define ACT_RELU 1
#define ACT_TANH 2

// CDNA5 async global->LDS copy path (ASYNCcnt), if the toolchain exposes it.
#if __has_builtin(__builtin_amdgcn_global_load_async_to_lds_b32) && \
    __has_builtin(__builtin_amdgcn_s_wait_asynccnt)
#define HAVE_ASYNC_LDS 1
#else
#define HAVE_ASYNC_LDS 0
#endif

// float -> bf16 bits, round-to-nearest-even
__device__ __forceinline__ unsigned short f2bf(float f) {
  unsigned int u = __float_as_uint(f);
  unsigned int r = u + 0x7FFFu + ((u >> 16) & 1u);
  if ((u & 0x7F800000u) == 0x7F800000u) r = u;   // inf/nan: truncate
  return (unsigned short)(r >> 16);
}

// ---------------------------------------------------------------------------
// Generic batched GEMM:  Y[b, m, n] = act( sum_k W[m,k] * X[b, k, n] + bias[m] )
//   W accessed as W[m*wRowStride + k*wColStride]  (covers the strided
//   ConvTranspose2d attn2 weights).
//   X row k of batch b at X + b*xBatch + k*xRowStride.
//
// Block: 256 threads = 8 waves; block tile 64(M) x 64(N); K-step 32.
// Wave (wm,wn) owns two 16x16 tiles: rows wm*16 and wm*16+32, cols wn*16,
// sharing one B fragment -> 2x v_wmma_f32_16x16x32_bf16 per K-step.
//
// LDS layout is fragment-friendly: a lane's 16 bf16 A elements are
//   K = {half*8..half*8+7} and {16+half*8..16+half*8+7}   (ISA 7.12.2)
// i.e. two contiguous 16B runs of As[m][*] -> two ds_load_b128 per fragment.
// B tile is staged transposed (BsT[n][k]) so B fragments read the same way.
// Row pitch 40 u16 (80B) keeps the 16B alignment and staggers banks.
// ---------------------------------------------------------------------------
template <int ACT, bool INRELU>
__global__ __launch_bounds__(256) void gemm_bias_act(
    const float* __restrict__ W, int wRowStride, int wColStride,
    const float* __restrict__ bias,
    const float* __restrict__ X, long xBatch, int xRowStride,
    float* __restrict__ Y, long yBatch,
    int M, int K, int N) {
  const int b  = blockIdx.z;
  const int m0 = blockIdx.y * 64;
  const int n0 = blockIdx.x * 64;
  const float* Xb = X + (long)b * xBatch;
  float* Yb = Y + (long)b * yBatch;

  const int t    = threadIdx.x;
  const int lane = t & 31;
  const int wave = t >> 5;     // 0..7
  const int wm   = wave >> 2;  // 0..1
  const int wn   = wave & 3;   // 0..3
  const int half = lane >> 4;  // 0/1
  const int lm   = lane & 15;

  __shared__ unsigned short As[64][40];   // [m][k] bf16 bits
  __shared__ unsigned short BsT[64][40];  // [n][k] bf16 bits (transposed tile)

  f32x8 acc0 = {};
  f32x8 acc1 = {};

  const bool wvec = (wColStride == 1) && ((wRowStride & 3) == 0);
  const bool xvec = ((xRowStride & 3) == 0);

  for (int kt = 0; kt < K; kt += 32) {
    // ---- stage A tile (64 x 32): 512 float4 groups, 2 per thread ----
    for (int i = t; i < 64 * 8; i += 256) {
      int mm = i >> 3, c4 = (i & 7) * 4;
      int gm = m0 + mm, gk = kt + c4;
      float v0 = 0.f, v1 = 0.f, v2 = 0.f, v3 = 0.f;
      if (gm < M) {
        if (wvec && gk + 4 <= K) {
          float4 f = *(const float4*)(W + (long)gm * wRowStride + gk);
          v0 = f.x; v1 = f.y; v2 = f.z; v3 = f.w;
        } else {
          const float* wr = W + (long)gm * wRowStride;
          if (gk + 0 < K) v0 = wr[(long)(gk + 0) * wColStride];
          if (gk + 1 < K) v1 = wr[(long)(gk + 1) * wColStride];
          if (gk + 2 < K) v2 = wr[(long)(gk + 2) * wColStride];
          if (gk + 3 < K) v3 = wr[(long)(gk + 3) * wColStride];
        }
      }
      u16x4 pk = {f2bf(v0), f2bf(v1), f2bf(v2), f2bf(v3)};
      *(u16x4*)&As[mm][c4] = pk;          // 8B aligned -> ds_store_b64
    }
    // ---- stage B tile (32 x 64) transposed: 512 float4 groups ----
    for (int i = t; i < 32 * 16; i += 256) {
      int kk = i >> 4, c4 = (i & 15) * 4;
      int gk = kt + kk, gn = n0 + c4;
      float v0 = 0.f, v1 = 0.f, v2 = 0.f, v3 = 0.f;
      if (gk < K) {
        const float* xr = Xb + (long)gk * xRowStride;
        if (xvec && gn + 4 <= N) {
          float4 f = *(const float4*)(xr + gn);
          v0 = f.x; v1 = f.y; v2 = f.z; v3 = f.w;
        } else {
          if (gn + 0 < N) v0 = xr[gn + 0];
          if (gn + 1 < N) v1 = xr[gn + 1];
          if (gn + 2 < N) v2 = xr[gn + 2];
          if (gn + 3 < N) v3 = xr[gn + 3];
        }
        if (INRELU) {
          v0 = fmaxf(v0, 0.f); v1 = fmaxf(v1, 0.f);
          v2 = fmaxf(v2, 0.f); v3 = fmaxf(v3, 0.f);
        }
      }
      BsT[c4 + 0][kk] = f2bf(v0);
      BsT[c4 + 1][kk] = f2bf(v1);
      BsT[c4 + 2][kk] = f2bf(v2);
      BsT[c4 + 3][kk] = f2bf(v3);
    }
    // prefetch next X K-tile into L2 (global_prefetch_b8)
    if (kt + 32 < K) {
      int pk = kt + 32 + (t >> 6) * 8;
      if (pk < K && n0 + (t & 63) < N)
        __builtin_prefetch(&Xb[(long)pk * xRowStride + n0 + (t & 63)], 0, 0);
    }
    __syncthreads();

    // ---- fragments: 2x ds_load_b128 each ----
    u16x8 a0lo = *(const u16x8*)&As[wm * 16 + lm][half * 8];
    u16x8 a0hi = *(const u16x8*)&As[wm * 16 + lm][16 + half * 8];
    u16x8 a1lo = *(const u16x8*)&As[wm * 16 + 32 + lm][half * 8];
    u16x8 a1hi = *(const u16x8*)&As[wm * 16 + 32 + lm][16 + half * 8];
    u16x8 blo  = *(const u16x8*)&BsT[wn * 16 + lm][half * 8];
    u16x8 bhi  = *(const u16x8*)&BsT[wn * 16 + lm][16 + half * 8];
    u16x16 au0 = __builtin_shufflevector(a0lo, a0hi, 0, 1, 2, 3, 4, 5, 6, 7,
                                         8, 9, 10, 11, 12, 13, 14, 15);
    u16x16 au1 = __builtin_shufflevector(a1lo, a1hi, 0, 1, 2, 3, 4, 5, 6, 7,
                                         8, 9, 10, 11, 12, 13, 14, 15);
    u16x16 bu  = __builtin_shufflevector(blo, bhi, 0, 1, 2, 3, 4, 5, 6, 7,
                                         8, 9, 10, 11, 12, 13, 14, 15);
    bf16x16 a0 = __builtin_bit_cast(bf16x16, au0);
    bf16x16 a1 = __builtin_bit_cast(bf16x16, au1);
    bf16x16 bf = __builtin_bit_cast(bf16x16, bu);
    acc0 = __builtin_amdgcn_wmma_f32_16x16x32_bf16(false, a0, false, bf,
                                                   (short)0, acc0, false, false);
    acc1 = __builtin_amdgcn_wmma_f32_16x16x32_bf16(false, a1, false, bf,
                                                   (short)0, acc1, false, false);
    __syncthreads();
  }

  // D layout: VGPR r -> row r (lanes 0-15) / r+8 (lanes 16-31); col = lane&15.
  const int col = n0 + wn * 16 + lm;
  if (col < N) {
#pragma unroll
    for (int r = 0; r < 8; ++r) {
      int row = m0 + wm * 16 + half * 8 + r;
      if (row < M) {
        float v = acc0[r] + bias[row];
        if (ACT == ACT_RELU) v = fmaxf(v, 0.f);
        if (ACT == ACT_TANH) v = tanhf(v);
        Yb[(long)row * N + col] = v;
      }
    }
#pragma unroll
    for (int r = 0; r < 8; ++r) {
      int row = m0 + 32 + wm * 16 + half * 8 + r;
      if (row < M) {
        float v = acc1[r] + bias[row];
        if (ACT == ACT_RELU) v = fmaxf(v, 0.f);
        if (ACT == ACT_TANH) v = tanhf(v);
        Yb[(long)row * N + col] = v;
      }
    }
  }
}

// ---------------------------------------------------------------------------
// KNN k=20 (self-KNN over pos (B,3,N)), brute force with LDS tiling.
// Position tiles are staged with GLOBAL_LOAD_ASYNC_TO_LDS (ASYNCcnt) when the
// toolchain exposes the builtin: pure global->LDS copy, no register hop.
// idxOut: (B, N, 20) int.
// ---------------------------------------------------------------------------
__global__ __launch_bounds__(128) void knn20_kernel(
    const float* __restrict__ pos, int N, int* __restrict__ idxOut) {
  const int b = blockIdx.y;
  const int q = blockIdx.x * 128 + threadIdx.x;
  const float* P = pos + (long)b * 3 * N;
  float qx = 0.f, qy = 0.f, qz = 0.f;
  if (q < N) { qx = P[q]; qy = P[N + q]; qz = P[2 * N + q]; }
  float bd[20];
  int bi[20];
#pragma unroll
  for (int i = 0; i < 20; ++i) { bd[i] = 3.0e38f; bi[i] = 0; }
  __shared__ float sx[128], sy[128], sz[128];
  for (int tile = 0; tile < N; tile += 128) {
    int j = tile + threadIdx.x;
#if HAVE_ASYNC_LDS
    if (j < N) {
      typedef __attribute__((address_space(1))) int GI;
      typedef __attribute__((address_space(3))) int LI;
      __builtin_amdgcn_global_load_async_to_lds_b32(
          (GI*)(P + j), (LI*)&sx[threadIdx.x], 0, 0);
      __builtin_amdgcn_global_load_async_to_lds_b32(
          (GI*)(P + N + j), (LI*)&sy[threadIdx.x], 0, 0);
      __builtin_amdgcn_global_load_async_to_lds_b32(
          (GI*)(P + 2 * N + j), (LI*)&sz[threadIdx.x], 0, 0);
    } else {
      sx[threadIdx.x] = 1e30f; sy[threadIdx.x] = 1e30f; sz[threadIdx.x] = 1e30f;
    }
    __builtin_amdgcn_s_wait_asynccnt(0);
    __syncthreads();
#else
    if (j < N) { sx[threadIdx.x] = P[j]; sy[threadIdx.x] = P[N + j]; sz[threadIdx.x] = P[2 * N + j]; }
    else       { sx[threadIdx.x] = 1e30f; sy[threadIdx.x] = 1e30f; sz[threadIdx.x] = 1e30f; }
    __syncthreads();
#endif
    for (int jj = 0; jj < 128; ++jj) {
      float dx = qx - sx[jj], dy = qy - sy[jj], dz = qz - sz[jj];
      float d = dx * dx + dy * dy + dz * dz;
      if (d < bd[19]) {
        int p = 19;
        while (p > 0 && bd[p - 1] > d) { bd[p] = bd[p - 1]; bi[p] = bi[p - 1]; --p; }
        bd[p] = d; bi[p] = tile + jj;
      }
    }
    __syncthreads();
  }
  if (q < N) {
    long o = ((long)b * N + q) * 20;
    for (int i = 0; i < 20; ++i) idxOut[o + i] = bi[i];
  }
}

// ---------------------------------------------------------------------------
// KNN k=3 of queries (pcd_prev) vs refs (seed) + inverse-distance weights.
// ---------------------------------------------------------------------------
__global__ void knn3_kernel(const float* __restrict__ qpos,
                            const float* __restrict__ rpos, int N, int NS,
                            int* __restrict__ idx3, float* __restrict__ w3) {
  const int b = blockIdx.y;
  const int q = blockIdx.x * blockDim.x + threadIdx.x;
  if (q >= N) return;
  const float* Qp = qpos + (long)b * 3 * N;
  const float* Rp = rpos + (long)b * 3 * NS;
  float qx = Qp[q], qy = Qp[N + q], qz = Qp[2 * N + q];
  float d0 = 3e38f, d1 = 3e38f, d2 = 3e38f;
  int i0 = 0, i1 = 0, i2 = 0;
  for (int j = 0; j < NS; ++j) {
    float dx = qx - Rp[j], dy = qy - Rp[NS + j], dz = qz - Rp[2 * NS + j];
    float d = dx * dx + dy * dy + dz * dz;
    if (d < d0)      { d2 = d1; i2 = i1; d1 = d0; i1 = i0; d0 = d; i0 = j; }
    else if (d < d1) { d2 = d1; i2 = i1; d1 = d;  i1 = j; }
    else if (d < d2) { d2 = d;  i2 = j; }
  }
  float w0 = 1.f / (d0 + 1e-8f), w1 = 1.f / (d1 + 1e-8f), w2 = 1.f / (d2 + 1e-8f);
  float s = w0 + w1 + w2;
  long o = ((long)b * N + q) * 3;
  idx3[o] = i0; idx3[o + 1] = i1; idx3[o + 2] = i2;
  w3[o] = w0 / s; w3[o + 1] = w1 / s; w3[o + 2] = w2 / s;
}

// feat_up[b,c,n] = sum_j w3[j] * seed_feat[b,c,idx3[j]]
__global__ void interp_kernel(const float* __restrict__ sf,
                              const int* __restrict__ idx3,
                              const float* __restrict__ w3,
                              float* __restrict__ out, int C, int N, int NS,
                              long total) {
  long i = blockIdx.x * (long)blockDim.x + threadIdx.x;
  if (i >= total) return;
  int n = (int)(i % N);
  long t = i / N;
  int c = (int)(t % C);
  int b = (int)(t / C);
  long o = ((long)b * N + n) * 3;
  const float* f = sf + ((long)b * C + c) * NS;
  out[i] = w3[o] * f[idx3[o]] + w3[o + 1] * f[idx3[o + 1]] + w3[o + 2] * f[idx3[o + 2]];
}

// gmax[b,c] = max_n x[b,c,n]
__global__ void rowmax_kernel(const float* __restrict__ x, float* __restrict__ out,
                              int C, int N) {
  int b = blockIdx.y;
  int c = blockIdx.x * blockDim.x + threadIdx.x;
  if (c >= C) return;
  const float* p = x + ((long)b * C + c) * N;
  float m = -3e38f;
  for (int n = 0; n < N; ++n) m = fmaxf(m, p[n]);
  out[b * C + c] = m;
}

// feat_1 concat: [f1(128) | gmax bcast(128) | gcn bcast(128) | feat_up(128)]
__global__ void pack512_kernel(const float* __restrict__ f1,
                               const float* __restrict__ gmax,
                               const float* __restrict__ gcn,
                               const float* __restrict__ fup,
                               float* __restrict__ out, int N, long total) {
  long i = blockIdx.x * (long)blockDim.x + threadIdx.x;
  if (i >= total) return;
  int n = (int)(i % N);
  long t = i / N;
  int c = (int)(t % 512);
  int b = (int)(t / 512);
  float v;
  if (c < 128)      v = f1[((long)b * 128 + c) * N + n];
  else if (c < 256) v = gmax[b * 128 + (c - 128)];
  else if (c < 384) v = gcn[b * 128 + (c - 256)];
  else              v = fup[((long)b * 128 + (c - 384)) * N + n];
  out[i] = v;
}

// concat channels: out(B,C1+C2,N) = [a(B,C1,N) | b2(B,C2,N/rep2) repeated]
__global__ void pack2_kernel(const float* __restrict__ a,
                             const float* __restrict__ b2,
                             float* __restrict__ out, int C1, int C2, int N,
                             int rep2, long total) {
  long i = blockIdx.x * (long)blockDim.x + threadIdx.x;
  if (i >= total) return;
  int n = (int)(i % N);
  long t = i / N;
  int c = (int)(t % (C1 + C2));
  int b = (int)(t / (C1 + C2));
  float v;
  if (c < C1) v = a[((long)b * C1 + c) * N + n];
  else        v = b2[((long)b * C2 + (c - C1)) * (N / rep2) + n / rep2];
  out[i] = v;
}

// pos_rel[b,d,(n,k)] = pos[b,d,n] - pos[b,d,idx[b,n,k]]
__global__ void posrel_kernel(const float* __restrict__ pos,
                              const int* __restrict__ idx,
                              float* __restrict__ out, int N, int KNN,
                              long total) {
  long i = blockIdx.x * (long)blockDim.x + threadIdx.x;
  if (i >= total) return;
  int k = (int)(i % KNN);
  long t = i / KNN;
  int n = (int)(t % N);
  long t2 = t / N;
  int d = (int)(t2 % 3);
  int b = (int)(t2 / 3);
  int j = idx[((long)b * N + n) * KNN + k];
  const float* P = pos + (long)b * 3 * N;
  out[i] = P[(long)d * N + n] - P[(long)d * N + j];
}

// hin[b,c,(n,k)] = (q[n]-k[idx]) + pe[(n,k)] + (uf[n]-uf[idx])
__global__ void hin_kernel(const float* __restrict__ q,
                           const float* __restrict__ kf,
                           const float* __restrict__ pe,
                           const float* __restrict__ uf,
                           const int* __restrict__ idx, float* __restrict__ out,
                           int C, int N, int KNN, long total) {
  long i = blockIdx.x * (long)blockDim.x + threadIdx.x;
  if (i >= total) return;
  int k = (int)(i % KNN);
  long t = i / KNN;
  int n = (int)(t % N);
  long t2 = t / N;
  int c = (int)(t2 % C);
  int b = (int)(t2 / C);
  int j = idx[((long)b * N + n) * KNN + k];
  long rb = ((long)b * C + c) * N;
  out[i] = (q[rb + n] - kf[rb + j]) + pe[i] + (uf[rb + n] - uf[rb + j]);
}

// BatchNorm statistics over (B, spatial) per channel
__global__ __launch_bounds__(256) void bnstats_kernel(
    const float* __restrict__ x, float* __restrict__ mean,
    float* __restrict__ var, int C, long NK, int B) {
  int c = blockIdx.x;
  float s = 0.f, s2 = 0.f;
  long tot = (long)B * NK;
  for (long i = threadIdx.x; i < tot; i += 256) {
    long bb = i / NK, n = i % NK;
    float v = x[(bb * C + c) * NK + n];
    s += v; s2 += v * v;
  }
  __shared__ float sh[256], sh2[256];
  sh[threadIdx.x] = s; sh2[threadIdx.x] = s2;
  __syncthreads();
  for (int off = 128; off > 0; off >>= 1) {
    if (threadIdx.x < off) {
      sh[threadIdx.x] += sh[threadIdx.x + off];
      sh2[threadIdx.x] += sh2[threadIdx.x + off];
    }
    __syncthreads();
  }
  if (threadIdx.x == 0) {
    float cnt = (float)tot;
    float m = sh[0] / cnt;
    mean[c] = m;
    var[c] = sh2[0] / cnt - m * m;
  }
}

// in-place BN apply (+ optional relu)
__global__ void bnapply_kernel(float* __restrict__ x,
                               const float* __restrict__ mean,
                               const float* __restrict__ var,
                               const float* __restrict__ g,
                               const float* __restrict__ beta, int C, long NK,
                               int relu, long total) {
  long i = blockIdx.x * (long)blockDim.x + threadIdx.x;
  if (i >= total) return;
  int c = (int)((i / NK) % C);
  float v = (x[i] - mean[c]) * rsqrtf(var[c] + 1e-5f) * g[c] + beta[c];
  if (relu) v = fmaxf(v, 0.f);
  x[i] = v;
}

// interleave attn2-up result: dst[b,c,(n*U+r),k] = src[b,c,(n,k)]
__global__ void interleave_kernel(const float* __restrict__ src,
                                  float* __restrict__ dst, int C, int N,
                                  int KNN, int U, int r, long total) {
  long i = blockIdx.x * (long)blockDim.x + threadIdx.x;
  if (i >= total) return;
  int k = (int)(i % KNN);
  long t = i / KNN;
  int n = (int)(t % N);
  long t2 = t / N;
  int c = (int)(t2 % C);
  int b = (int)(t2 / C);
  dst[(((long)b * C + c) * ((long)N * U) + (long)n * U + r) * KNN + k] = src[i];
}

// softmax over K=20 + fused val gather/aggregate:
//  val[k] = v[idx] + pe[(n,k)] + (uf[n]-uf[idx]);  agg[b,c,n2] = sum p_k val_k
__global__ void softmax_agg_kernel(const float* __restrict__ attn,
                                   const float* __restrict__ vf,
                                   const float* __restrict__ pe,
                                   const float* __restrict__ uf,
                                   const int* __restrict__ idx,
                                   float* __restrict__ agg, int C, int N,
                                   int KNN, int U, long total) {
  long i = blockIdx.x * (long)blockDim.x + threadIdx.x;
  if (i >= total) return;
  int n2 = (int)(i % ((long)N * U));
  long t = i / ((long)N * U);
  int c = (int)(t % C);
  int b = (int)(t / C);
  int n = n2 / U;
  const float* al = attn + i * KNN;
  float mx = -3e38f;
  for (int k = 0; k < KNN; ++k) mx = fmaxf(mx, al[k]);
  long rb = ((long)b * C + c) * N;
  long peb = ((long)b * C + c) * (long)N * KNN + (long)n * KNN;
  long ib = ((long)b * N + n) * KNN;
  float ufn = uf[rb + n];
  float se = 0.f, acc = 0.f;
  for (int k = 0; k < KNN; ++k) {
    float p = __expf(al[k] - mx);
    int j = idx[ib + k];
    float val = vf[rb + j] + pe[peb + k] + (ufn - uf[rb + j]);
    se += p;
    acc += p * val;
  }
  agg[i] = acc / se;
}

// y[b,c,n2] += id[b,c,n2/U]   (residual with nearest upsample)
__global__ void addrepeat_kernel(float* __restrict__ y,
                                 const float* __restrict__ id, int C, int N2,
                                 int U, long total) {
  long i = blockIdx.x * (long)blockDim.x + threadIdx.x;
  if (i >= total) return;
  int n2 = (int)(i % N2);
  long t = i / N2;
  int c = (int)(t % C);
  int b = (int)(t / C);
  y[i] += id[((long)b * C + c) * (N2 / U) + n2 / U];
}

__global__ void eltadd_kernel(const float* __restrict__ a,
                              const float* __restrict__ b,
                              float* __restrict__ c, long total) {
  long i = blockIdx.x * (long)blockDim.x + threadIdx.x;
  if (i < total) c[i] = a[i] + b[i];
}

// pcd_child[b,d,n2] = pcd_prev[b,d,n2/U] + delta[b,d,n2]
__global__ void final_pcd_kernel(const float* __restrict__ prev,
                                 const float* __restrict__ delta,
                                 float* __restrict__ out, int N, int U,
                                 long total) {
  long i = blockIdx.x * (long)blockDim.x + threadIdx.x;
  if (i >= total) return;
  int n2 = (int)(i % ((long)N * U));
  long t = i / ((long)N * U);
  int d = (int)(t % 3);
  int b = (int)(t / 3);
  out[i] = prev[((long)b * 3 + d) * N + n2 / U] + delta[i];
}

// ---------------------------------------------------------------------------
// Host-side helpers
// ---------------------------------------------------------------------------
static inline void run_gemm(hipStream_t s, const float* W, int wrs, int wcs,
                            const float* bias, const float* X, long xbs,
                            int xrs, float* Y, long ybs, int M, int K, int N,
                            int B, int act, bool inrelu) {
  dim3 g((N + 63) / 64, (M + 63) / 64, B), blk(256);
  if (inrelu)
    gemm_bias_act<ACT_RELU, true><<<g, blk, 0, s>>>(W, wrs, wcs, bias, X, xbs, xrs, Y, ybs, M, K, N);
  else if (act == ACT_RELU)
    gemm_bias_act<ACT_RELU, false><<<g, blk, 0, s>>>(W, wrs, wcs, bias, X, xbs, xrs, Y, ybs, M, K, N);
  else if (act == ACT_TANH)
    gemm_bias_act<ACT_TANH, false><<<g, blk, 0, s>>>(W, wrs, wcs, bias, X, xbs, xrs, Y, ybs, M, K, N);
  else
    gemm_bias_act<ACT_NONE, false><<<g, blk, 0, s>>>(W, wrs, wcs, bias, X, xbs, xrs, Y, ybs, M, K, N);
}

static inline long cdiv(long a, long b) { return (a + b - 1) / b; }

struct UT {  // one up-transformer's params (flatten order: sorted dict keys)
  const float *attn1_w, *attn1_b, *attn2_w, *attn2_b, *attn_bn_g, *attn_bn_b;
  const float *end_w, *end_b, *key_w, *key_b;
  const float *v_c1_w, *v_c1_b, *v_c2_w, *v_c2_b, *v_sc_w, *v_sc_b;
  const float *pos1_w, *pos1_b, *pos2_w, *pos2_b, *pos_bn_g, *pos_bn_b;
  const float *query_w, *query_b, *upfeat_w, *upfeat_b, *value_w, *value_b;
};

// One up-transformer forward.  up==0 -> no upsample; else up factor U.
static void run_uptrans(hipStream_t s, const UT& p, const float* pos,
                        const float* keyf, const float* qryf, const float* upf,
                        int up, int N, int B,
                        float* cat2, float* tv, float* va, float* value,
                        float* kf, float* qf, float* vf, float* uff,
                        float* posrel, float* pe, float* hin, float* big,
                        float* logits, float* attn2tmp, float* agg,
                        const int* idx20, float* bnm, float* bnv,
                        float* yout) {
  const int KNN = 20;
  const long NK = (long)N * KNN;
  const int U = up ? up : 1;
  const int N2 = N * U;

  long tot;
  // cat2 = [keyf ; qryf]
  tot = (long)B * 256 * N;
  pack2_kernel<<<cdiv(tot, 256), 256, 0, s>>>(keyf, qryf, cat2, 128, 128, N, 1, tot);
  // value = mlp_res(cat2)  (identity of this transformer)
  run_gemm(s, p.v_c1_w, 256, 1, p.v_c1_b, cat2, 256L * N, N, tv, 128L * N, 128, 256, N, B, ACT_RELU, false);
  run_gemm(s, p.v_c2_w, 128, 1, p.v_c2_b, tv, 128L * N, N, va, 128L * N, 128, 128, N, B, ACT_NONE, false);
  run_gemm(s, p.v_sc_w, 256, 1, p.v_sc_b, cat2, 256L * N, N, value, 128L * N, 128, 256, N, B, ACT_NONE, false);
  tot = (long)B * 128 * N;
  eltadd_kernel<<<cdiv(tot, 256), 256, 0, s>>>(va, value, value, tot);
  // k, q, v, upfeat projections (dim 64)
  run_gemm(s, p.key_w, 128, 1, p.key_b, keyf, 128L * N, N, kf, 64L * N, 64, 128, N, B, ACT_NONE, false);
  run_gemm(s, p.query_w, 128, 1, p.query_b, qryf, 128L * N, N, qf, 64L * N, 64, 128, N, B, ACT_NONE, false);
  run_gemm(s, p.value_w, 128, 1, p.value_b, value, 128L * N, N, vf, 64L * N, 64, 128, N, B, ACT_NONE, false);
  run_gemm(s, p.upfeat_w, 128, 1, p.upfeat_b, upf, 128L * N, N, uff, 64L * N, 64, 128, N, B, ACT_NONE, false);
  // positional encoding: pe = conv2(relu(bn(conv1(pos_rel))))
  tot = (long)B * 3 * NK;
  posrel_kernel<<<cdiv(tot, 256), 256, 0, s>>>(pos, idx20, posrel, N, KNN, tot);
  run_gemm(s, p.pos1_w, 3, 1, p.pos1_b, posrel, 3L * NK, (int)NK, attn2tmp, 64L * NK, 64, 3, (int)NK, B, ACT_NONE, false);
  bnstats_kernel<<<64, 256, 0, s>>>(attn2tmp, bnm, bnv, 64, NK, B);
  tot = (long)B * 64 * NK;
  bnapply_kernel<<<cdiv(tot, 256), 256, 0, s>>>(attn2tmp, bnm, bnv, p.pos_bn_g, p.pos_bn_b, 64, NK, 1, tot);
  run_gemm(s, p.pos2_w, 64, 1, p.pos2_b, attn2tmp, 64L * NK, (int)NK, pe, 64L * NK, 64, 64, (int)NK, B, ACT_NONE, false);
  // h input = qk_rel + pe + uf_rel
  hin_kernel<<<cdiv(tot, 256), 256, 0, s>>>(qf, kf, pe, uff, idx20, hin, 64, N, KNN, tot);
  // attn1 + bn + relu  -> big (B,256,NK)
  run_gemm(s, p.attn1_w, 64, 1, p.attn1_b, hin, 64L * NK, (int)NK, big, 256L * NK, 256, 64, (int)NK, B, ACT_NONE, false);
  bnstats_kernel<<<256, 256, 0, s>>>(big, bnm, bnv, 256, NK, B);
  tot = (long)B * 256 * NK;
  bnapply_kernel<<<cdiv(tot, 256), 256, 0, s>>>(big, bnm, bnv, p.attn_bn_g, p.attn_bn_b, 256, NK, 1, tot);
  // attn2 -> logits (B,64,N2,KNN)
  if (!up) {
    run_gemm(s, p.attn2_w, 256, 1, p.attn2_b, big, 256L * NK, (int)NK, logits, 64L * NK, 64, 256, (int)NK, B, ACT_NONE, false);
  } else {
    // ConvTranspose2d weights wt (C=256, O=64, R=U): W_r[o][c] = wt[c*64*U + o*U + r]
    for (int r = 0; r < U; ++r) {
      run_gemm(s, p.attn2_w + r, U, 64 * U, p.attn2_b, big, 256L * NK, (int)NK,
               attn2tmp, 64L * NK, 64, 256, (int)NK, B, ACT_NONE, false);
      tot = (long)B * 64 * NK;
      interleave_kernel<<<cdiv(tot, 256), 256, 0, s>>>(attn2tmp, logits, 64, N, KNN, U, r, tot);
    }
  }
  // softmax over K + fused value gather + aggregate
  tot = (long)B * 64 * N2;
  softmax_agg_kernel<<<cdiv(tot, 256), 256, 0, s>>>(logits, vf, pe, uff, idx20, agg, 64, N, KNN, U, tot);
  // end conv + residual identity (repeated for up)
  run_gemm(s, p.end_w, 64, 1, p.end_b, agg, 64L * N2, N2, yout, 128L * N2, 128, 64, N2, B, ACT_NONE, false);
  tot = (long)B * 128 * N2;
  addrepeat_kernel<<<cdiv(tot, 256), 256, 0, s>>>(yout, value, 128, N2, U, tot);
}

// ---------------------------------------------------------------------------
// kernel_launch
// ---------------------------------------------------------------------------
extern "C" void kernel_launch(void* const* d_in, const int* in_sizes, int n_in,
                              void* d_out, int out_size, void* d_ws,
                              size_t ws_size, hipStream_t stream) {
  const int B = 4, N = 2048, NS = 512, KNN = 20, UPF = 2;
  const int N2 = N * UPF;
  const long NK = (long)N * KNN;

  const float* pcd   = (const float*)d_in[0];   // (B,3,N)
  const float* gcn   = (const float*)d_in[1];   // (B,128,1)
  const float* seed  = (const float*)d_in[2];   // (B,3,NS)
  const float* seedf = (const float*)d_in[3];   // (B,128,NS)
  const float* Kprev = (const float*)d_in[4];   // (B,128,N)

  // params flattened: dicts in sorted-key order, lists/tuples in order.
  int pi = 5;
  auto nx = [&]() { return (const float*)d_in[pi++]; };
  const float *mlp1_w0 = nx(), *mlp1_b0 = nx(), *mlp1_w1 = nx(), *mlp1_b1 = nx();
  const float *mlp2_w0 = nx(), *mlp2_b0 = nx(), *mlp2_w1 = nx(), *mlp2_b1 = nx();
  const float *mld_w0 = nx(), *mld_b0 = nx(), *mld_w1 = nx(), *mld_b1 = nx();
  const float *mdf_c1w = nx(), *mdf_c1b = nx(), *mdf_c2w = nx(), *mdf_c2b = nx();
  const float *mdf_scw = nx(), *mdf_scb = nx();
  UT ut1, ut2;
  auto loadUT = [&](UT& u) {
    u.attn1_w = nx(); u.attn1_b = nx(); u.attn2_w = nx(); u.attn2_b = nx();
    u.attn_bn_g = nx(); u.attn_bn_b = nx(); u.end_w = nx(); u.end_b = nx();
    u.key_w = nx(); u.key_b = nx();
    u.v_c1_w = nx(); u.v_c1_b = nx(); u.v_c2_w = nx(); u.v_c2_b = nx();
    u.v_sc_w = nx(); u.v_sc_b = nx();
    u.pos1_w = nx(); u.pos1_b = nx(); u.pos2_w = nx(); u.pos2_b = nx();
    u.pos_bn_g = nx(); u.pos_bn_b = nx();
    u.query_w = nx(); u.query_b = nx(); u.upfeat_w = nx(); u.upfeat_b = nx();
    u.value_w = nx(); u.value_b = nx();
  };
  loadUT(ut1);
  loadUT(ut2);

  // workspace bump allocator
  char* wsp = (char*)d_ws;
  size_t off = 0;
  auto alloc = [&](size_t bytes) -> void* {
    void* p = wsp + off;
    off = (off + bytes + 255) & ~(size_t)255;
    return p;
  };
  int*   idx3    = (int*)alloc((size_t)B * N * 3 * 4);
  float* w3      = (float*)alloc((size_t)B * N * 3 * 4);
  float* featup  = (float*)alloc((size_t)B * 128 * N * 4);
  float* t64     = (float*)alloc((size_t)B * 64 * N * 4);
  float* feat1a  = (float*)alloc((size_t)B * 128 * N * 4);
  float* gmax    = (float*)alloc((size_t)B * 128 * 4);
  float* featcat = (float*)alloc((size_t)B * 512 * N * 4);
  float* t256    = (float*)alloc((size_t)B * 256 * N * 4);
  float* Q       = (float*)alloc((size_t)B * 128 * N * 4);
  int*   idx20   = (int*)alloc((size_t)B * N * KNN * 4);
  float* cat2    = (float*)alloc((size_t)B * 256 * N * 4);
  float* tv      = (float*)alloc((size_t)B * 128 * N * 4);
  float* va      = (float*)alloc((size_t)B * 128 * N * 4);
  float* value1  = (float*)alloc((size_t)B * 128 * N * 4);
  float* value2  = (float*)alloc((size_t)B * 128 * N * 4);
  float* kf      = (float*)alloc((size_t)B * 64 * N * 4);
  float* qf      = (float*)alloc((size_t)B * 64 * N * 4);
  float* vf      = (float*)alloc((size_t)B * 64 * N * 4);
  float* uff     = (float*)alloc((size_t)B * 64 * N * 4);
  float* posrel  = (float*)alloc((size_t)B * 3 * NK * 4);
  float* pe      = (float*)alloc((size_t)B * 64 * NK * 4);
  float* hin     = (float*)alloc((size_t)B * 64 * NK * 4);
  float* big     = (float*)alloc((size_t)B * 256 * NK * 4);
  float* attn2tmp= (float*)alloc((size_t)B * 64 * NK * 4);
  float* logits1 = (float*)alloc((size_t)B * 64 * NK * 4);
  float* logits2 = (float*)alloc((size_t)B * 64 * (size_t)N2 * KNN * 4);
  float* agg1    = (float*)alloc((size_t)B * 64 * N * 4);
  float* agg2    = (float*)alloc((size_t)B * 64 * N2 * 4);
  float* H       = (float*)alloc((size_t)B * 128 * N * 4);
  float* featch  = (float*)alloc((size_t)B * 128 * N2 * 4);
  float* cat3    = (float*)alloc((size_t)B * 256 * N2 * 4);
  float* kt1     = (float*)alloc((size_t)B * 128 * N2 * 4);
  float* kt2     = (float*)alloc((size_t)B * 128 * N2 * 4);
  float* t64b    = (float*)alloc((size_t)B * 64 * N2 * 4);
  float* delta   = (float*)alloc((size_t)B * 3 * N2 * 4);
  float* bnm     = (float*)alloc(256 * 4);
  float* bnv     = (float*)alloc(256 * 4);
  (void)ws_size; (void)in_sizes; (void)n_in; (void)out_size;

  float* out_pcd = (float*)d_out;                    // (B,3,N2)
  float* out_K   = out_pcd + (size_t)B * 3 * N2;     // (B,128,N2)

  long tot;
  // --- seed-feature KNN-3 interpolation ---
  knn3_kernel<<<dim3((N + 127) / 128, B), 128, 0, stream>>>(pcd, seed, N, NS, idx3, w3);
  tot = (long)B * 128 * N;
  interp_kernel<<<cdiv(tot, 256), 256, 0, stream>>>(seedf, idx3, w3, featup, 128, N, NS, tot);
  // --- mlp_1 ---
  run_gemm(stream, mlp1_w0, 3, 1, mlp1_b0, pcd, 3L * N, N, t64, 64L * N, 64, 3, N, B, ACT_RELU, false);
  run_gemm(stream, mlp1_w1, 64, 1, mlp1_b1, t64, 64L * N, N, feat1a, 128L * N, 128, 64, N, B, ACT_NONE, false);
  rowmax_kernel<<<dim3(1, B), 128, 0, stream>>>(feat1a, gmax, 128, N);
  tot = (long)B * 512 * N;
  pack512_kernel<<<cdiv(tot, 256), 256, 0, stream>>>(feat1a, gmax, gcn, featup, featcat, N, tot);
  // --- mlp_2 -> Q ---
  run_gemm(stream, mlp2_w0, 512, 1, mlp2_b0, featcat, 512L * N, N, t256, 256L * N, 256, 512, N, B, ACT_RELU, false);
  run_gemm(stream, mlp2_w1, 256, 1, mlp2_b1, t256, 256L * N, N, Q, 128L * N, 128, 256, N, B, ACT_NONE, false);
  // --- self-KNN (shared by both transformers) ---
  knn20_kernel<<<dim3((N + 127) / 128, B), 128, 0, stream>>>(pcd, N, idx20);
  // --- up_transformer 1 (no upsample) -> H ---
  run_uptrans(stream, ut1, pcd, Kprev, Q, featup, 0, N, B, cat2, tv, va, value1,
              kf, qf, vf, uff, posrel, pe, hin, big, logits1, attn2tmp, agg1,
              idx20, bnm, bnv, H);
  // --- up_transformer 2 (up=2) -> feat_child ---
  run_uptrans(stream, ut2, pcd, Kprev, H, featup, UPF, N, B, cat2, tv, va, value2,
              kf, qf, vf, uff, posrel, pe, hin, big, logits2, attn2tmp, agg2,
              idx20, bnm, bnv, featch);
  // --- K_curr = mlp_res([feat_child ; repeat(H,2)]) ---
  tot = (long)B * 256 * N2;
  pack2_kernel<<<cdiv(tot, 256), 256, 0, stream>>>(featch, H, cat3, 128, 128, N2, UPF, tot);
  run_gemm(stream, mdf_c1w, 256, 1, mdf_c1b, cat3, 256L * N2, N2, kt1, 128L * N2, 128, 256, N2, B, ACT_RELU, false);
  run_gemm(stream, mdf_c2w, 128, 1, mdf_c2b, kt1, 128L * N2, N2, kt2, 128L * N2, 128, 128, N2, B, ACT_NONE, false);
  run_gemm(stream, mdf_scw, 256, 1, mdf_scb, cat3, 256L * N2, N2, out_K, 128L * N2, 128, 256, N2, B, ACT_NONE, false);
  tot = (long)B * 128 * N2;
  eltadd_kernel<<<cdiv(tot, 256), 256, 0, stream>>>(kt2, out_K, out_K, tot);
  // --- delta = tanh(mlp_delta(relu(K_curr))) ; pcd_child ---
  run_gemm(stream, mld_w0, 128, 1, mld_b0, out_K, 128L * N2, N2, t64b, 64L * N2, 64, 128, N2, B, ACT_RELU, true);
  run_gemm(stream, mld_w1, 64, 1, mld_b1, t64b, 64L * N2, N2, delta, 3L * N2, 3, 64, N2, B, ACT_TANH, false);
  tot = (long)B * 3 * N2;
  final_pcd_kernel<<<cdiv(tot, 256), 256, 0, stream>>>(pcd, delta, out_pcd, N, UPF, tot);
}
